// DeatilAlignedModule_86784109182993
// MI455X (gfx1250) — matrix operand order repaired and checked by reading, not compiled, not run on hardware
//
#include <hip/hip_runtime.h>
#include <math.h>

#define B_   8
#define C_   64
#define LH_  128
#define LW_  128
#define HH_  64
#define HW_  64

typedef __bf16          v16bf __attribute__((ext_vector_type(16)));
typedef unsigned short  v16u  __attribute__((ext_vector_type(16)));
typedef unsigned short  v8u   __attribute__((ext_vector_type(8)));
typedef float           v8f   __attribute__((ext_vector_type(8)));

__device__ __forceinline__ unsigned short f2bf(float f) {
    unsigned int u = __float_as_uint(f);
    unsigned int r = u + 0x7FFFu + ((u >> 16) & 1u);   // RNE
    return (unsigned short)(r >> 16);
}
__device__ __forceinline__ float bf2f(unsigned short h) {
    return __uint_as_float(((unsigned int)h) << 16);
}

// Reference-faithful grid_sample tap: value read at clamped index, weight
// zeroed when the un-clamped tap falls outside.
__device__ __forceinline__ float bilin_sample(const float* __restrict__ img,
                                              int H, int W, float gx, float gy) {
    float x0f = floorf(gx), y0f = floorf(gy);
    float wx1 = gx - x0f,  wy1 = gy - y0f;
    float wx0 = 1.f - wx1, wy0 = 1.f - wy1;
    int x0 = (int)x0f, y0 = (int)y0f;
    int x1 = x0 + 1,   y1 = y0 + 1;
    float m00 = ((x0 >= 0) & (x0 < W) & (y0 >= 0) & (y0 < H)) ? 1.f : 0.f;
    float m10 = ((x1 >= 0) & (x1 < W) & (y0 >= 0) & (y0 < H)) ? 1.f : 0.f;
    float m01 = ((x0 >= 0) & (x0 < W) & (y1 >= 0) & (y1 < H)) ? 1.f : 0.f;
    float m11 = ((x1 >= 0) & (x1 < W) & (y1 >= 0) & (y1 < H)) ? 1.f : 0.f;
    int cx0 = min(max(x0, 0), W - 1), cx1 = min(max(x1, 0), W - 1);
    int cy0 = min(max(y0, 0), H - 1), cy1 = min(max(y1, 0), H - 1);
    return img[cy0 * W + cx0] * (wx0 * wy0 * m00)
         + img[cy0 * W + cx1] * (wx1 * wy0 * m10)
         + img[cy1 * W + cx0] * (wx0 * wy1 * m01)
         + img[cy1 * W + cx1] * (wx1 * wy1 * m11);
}

// ---------------------------------------------------------------------------
// K0: repack w_flow (256,64,3,3) f32 -> bf16 panel in WMMA A-FRAGMENT order:
//     A[chunk][mtile][lane][e], chunk = K-slab of 32 (tap-major K = tap*64+c).
//     K4 then loads fragments straight from global (L2-resident, no LDS).
//     ISA 16-bit A map inverted: lane half h, elem e -> kl = e<8 ? e+8h : e+8+8h
// ---------------------------------------------------------------------------
__global__ void k_prep_weights(const float* __restrict__ w_flow,
                               unsigned short* __restrict__ A) {
    int i = blockIdx.x * blockDim.x + threadIdx.x;       // 18*16*32*16 = 147456
    if (i >= 18 * 16 * 32 * 16) return;
    int e     = i & 15;
    int lane  = (i >> 4) & 31;
    int mtile = (i >> 9) & 15;
    int chunk = i >> 13;
    int half  = lane >> 4;
    int kloc  = (e < 8) ? (e + 8 * half) : (e + 8 + 8 * half);
    int k     = chunk * 32 + kloc;                       // 0..575
    int m     = mtile * 16 + (lane & 15);                // out channel
    int t = k >> 6, c = k & 63;                          // tap-major K
    A[i] = f2bf(w_flow[(m * 64 + c) * 9 + t]);
}

// ---------------------------------------------------------------------------
// K1: depthwise 3x3 on high -> hfeat (8,64,64,64)
// ---------------------------------------------------------------------------
__global__ void k_dw_high(const float* __restrict__ high,
                          const float* __restrict__ wc,
                          float* __restrict__ hfeat) {
    int i = blockIdx.x * blockDim.x + threadIdx.x;
    if (i >= B_ * C_ * HH_ * HW_) return;
    int x = i & 63, y = (i >> 6) & 63, c = (i >> 12) & 63, b = i >> 18;
    const float* img = high + (b * C_ + c) * HH_ * HW_;
    const float* w   = wc + c * 9;
    float acc = 0.f;
#pragma unroll
    for (int ky = 0; ky < 3; ++ky) {
        int yy = y + ky - 1;
        if (yy < 0 || yy >= HH_) continue;
#pragma unroll
        for (int kx = 0; kx < 3; ++kx) {
            int xx = x + kx - 1;
            if (xx < 0 || xx >= HW_) continue;
            acc = fmaf(img[yy * HW_ + xx], w[ky * 3 + kx], acc);
        }
    }
    hfeat[i] = acc;
}

// ---------------------------------------------------------------------------
// K2: per 128x128 pixel: depthwise-low conv + bilinear-upsampled hfeat,
//     store bf16 sum (GEMM input) + channel mean/max -> gate_in (B,4,128,128)
// ---------------------------------------------------------------------------
__global__ void k_fuse_pre(const float* __restrict__ low,
                           const float* __restrict__ wl,
                           const float* __restrict__ hfeat,
                           unsigned short* __restrict__ s_bf,
                           float* __restrict__ gate_in) {
    int i = blockIdx.x * blockDim.x + threadIdx.x;       // B*128*128
    if (i >= B_ * LH_ * LW_) return;
    int x = i & 127, y = (i >> 7) & 127, b = i >> 14;

    // jax.image.resize bilinear 64->128: src = dst*0.5 - 0.25, edge-clamped
    float sx = fminf(fmaxf(x * 0.5f - 0.25f, 0.f), (float)(HW_ - 1));
    float sy = fminf(fmaxf(y * 0.5f - 0.25f, 0.f), (float)(HH_ - 1));
    int   ix0 = (int)sx, iy0 = (int)sy;
    int   ix1 = min(ix0 + 1, HW_ - 1), iy1 = min(iy0 + 1, HH_ - 1);
    float fx = sx - ix0, fy = sy - iy0;
    float w00 = (1.f - fx) * (1.f - fy), w10 = fx * (1.f - fy);
    float w01 = (1.f - fx) * fy,         w11 = fx * fy;

    float sum_h = 0.f, sum_l = 0.f, max_h = -3.402823e38f, max_l = -3.402823e38f;
    for (int c = 0; c < C_; ++c) {
        const float* img = low + (b * C_ + c) * LH_ * LW_;
        const float* w   = wl + c * 9;
        float l = 0.f;
#pragma unroll
        for (int ky = 0; ky < 3; ++ky) {
            int yy = y + ky - 1;
            if (yy < 0 || yy >= LH_) continue;
#pragma unroll
            for (int kx = 0; kx < 3; ++kx) {
                int xx = x + kx - 1;
                if (xx < 0 || xx >= LW_) continue;
                l = fmaf(img[yy * LW_ + xx], w[ky * 3 + kx], l);
            }
        }
        const float* him = hfeat + (b * C_ + c) * HH_ * HW_;
        float h = him[iy0 * HW_ + ix0] * w00 + him[iy0 * HW_ + ix1] * w10
                + him[iy1 * HW_ + ix0] * w01 + him[iy1 * HW_ + ix1] * w11;

        sum_h += h; sum_l += l;
        max_h = fmaxf(max_h, h); max_l = fmaxf(max_l, l);
        s_bf[((b * C_ + c) * LH_ + y) * LW_ + x] = f2bf(l + h);
    }
    int pp = y * LW_ + x;
    gate_in[(b * 4 + 0) * (LH_ * LW_) + pp] = sum_h * (1.f / C_);
    gate_in[(b * 4 + 1) * (LH_ * LW_) + pp] = sum_l * (1.f / C_);
    gate_in[(b * 4 + 2) * (LH_ * LW_) + pp] = max_h;
    gate_in[(b * 4 + 3) * (LH_ * LW_) + pp] = max_l;
}

// ---------------------------------------------------------------------------
// K3: 3x3 conv (4 -> 1) + sigmoid -> gates (B,1,128,128)
// ---------------------------------------------------------------------------
__global__ void k_gate_conv(const float* __restrict__ gate_in,
                            const float* __restrict__ wg,
                            float* __restrict__ gates) {
    int i = blockIdx.x * blockDim.x + threadIdx.x;
    if (i >= B_ * LH_ * LW_) return;
    int x = i & 127, y = (i >> 7) & 127, b = i >> 14;
    float acc = 0.f;
#pragma unroll
    for (int ch = 0; ch < 4; ++ch) {
        const float* img = gate_in + (b * 4 + ch) * (LH_ * LW_);
        const float* w   = wg + ch * 9;
#pragma unroll
        for (int ky = 0; ky < 3; ++ky) {
            int yy = y + ky - 1;
            if (yy < 0 || yy >= LH_) continue;
#pragma unroll
            for (int kx = 0; kx < 3; ++kx) {
                int xx = x + kx - 1;
                if (xx < 0 || xx >= LW_) continue;
                acc = fmaf(img[yy * LW_ + xx], w[ky * 3 + kx], acc);
            }
        }
    }
    gates[b * (LH_ * LW_) + y * LW_ + x] = 1.f / (1.f + expf(-acc));
}

// ---------------------------------------------------------------------------
// B-chunk staging for K4: thread fills 8 CONTIGUOUS fragment slots
// (fixed pixel, 8 consecutive channels) -> 8 independent global_load_u16
// (coalesced across lanes, single wait) + one vector ds_store_b128.
// One bounds predicate per thread (halo pixel in/out of image).
// ---------------------------------------------------------------------------
__device__ __forceinline__ void stage_B(const unsigned short* __restrict__ S,
                                        unsigned short* __restrict__ dst,
                                        int b, int y, int chunk, int tid) {
    int t  = chunk >> 1;                  // tap 0..8
    int c0 = (chunk & 1) << 5;            // channel base 0/32
    int dy = t / 3 - 1, dx = t % 3 - 1;

    int f0 = tid << 3;                    // fragment-linear index, 8 per thread
    int e0 = f0 & 15;                     // 0 or 8
    int ln = (f0 >> 4) & 31;              // fragment lane
    int nt = f0 >> 9;                     // N-tile 0..7
    int n  = (nt << 4) + (ln & 15);       // pixel 0..127
    int ch = c0 + ((ln >> 4) << 3) + (e0 << 1);  // first of 8 consecutive channels

    int yy = y + dy, xx = n + dx;
    v8u v = (v8u){0, 0, 0, 0, 0, 0, 0, 0};
    if (yy >= 0 && yy < LH_ && xx >= 0 && xx < LW_) {
        const unsigned short* src = S + (((b * C_ + ch) * LH_ + yy) * LW_ + xx);
#pragma unroll
        for (int q = 0; q < 8; ++q)
            v[q] = src[q * (LH_ * LW_)];  // stride = one channel plane
    }
    *(v8u*)&dst[f0] = v;
}

// ---------------------------------------------------------------------------
// K4: implicit-GEMM flow conv (WMMA bf16, M=256 K=576 N=128/row) fused with
//     bilinear warps + gated blend. Flow never leaves LDS.
//     A fragments stream straight from global (pre-swizzled, L2-resident);
//     B patches double-buffered in LDS. Block = 512 threads = 16 waves.
// ---------------------------------------------------------------------------
__launch_bounds__(512, 1)
__global__ void k_flow_wmma_warp(const unsigned short* __restrict__ A,   // frag-order bf16
                                 const unsigned short* __restrict__ S,   // [B][C][128][128] bf16
                                 const float* __restrict__ low,
                                 const float* __restrict__ high,
                                 const float* __restrict__ gates,
                                 float* __restrict__ out) {
    // 64 KB LDS: B double buffer (2 x 8KB) unioned with flow bf16 panel.
    __shared__ __align__(32) unsigned short smem[32768];
    unsigned short* F = smem;             // flow bf16 [256 ch][128 px] (after GEMM)

    int blk = blockIdx.x;                 // B*LH blocks
    int y = blk & 127, b = blk >> 7;

    int tid  = threadIdx.x;
    int lane = tid & 31;
    int wave = tid >> 5;                  // 0..15
    int mt_base = (wave & 7) * 2;         // 2 M-tiles  (32 out-channels)
    int nt_base = (wave >> 3) * 4;        // 4 N-tiles  (64 pixels)

    v8f acc[2][4];
#pragma unroll
    for (int mi = 0; mi < 2; ++mi)
#pragma unroll
        for (int ni = 0; ni < 4; ++ni)
            acc[mi][ni] = (v8f){0.f, 0.f, 0.f, 0.f, 0.f, 0.f, 0.f, 0.f};

    // prologue: stage chunk 0 into buffer 0
    stage_B(S, smem, b, y, 0, tid);

    for (int chunk = 0; chunk < 18; ++chunk) {           // K = 18 * 32 = 576
        __syncthreads();
        // double-buffer bases by offset arithmetic (no LDS pointer arrays!)
        const unsigned short* cur = smem + ((chunk & 1) << 12);
        unsigned short* nxt       = smem + (((chunk + 1) & 1) << 12);

        // A fragments straight from global (coalesced b128 pairs, L2 hits)
        const unsigned short* Ac = A + ((size_t)chunk << 13);   // 16*32*16 per chunk
        if (chunk < 17)
            __builtin_prefetch(&A[((size_t)(chunk + 1) << 13) + (tid << 4)], 0, 1);

        v16bf afr[2];
#pragma unroll
        for (int mi = 0; mi < 2; ++mi)
            afr[mi] = __builtin_bit_cast(v16bf,
                *(const v16u*)&Ac[(((mt_base + mi) << 5) + lane) << 4]);

        v16bf bfr[4];
#pragma unroll
        for (int ni = 0; ni < 4; ++ni)
            bfr[ni] = __builtin_bit_cast(v16bf,
                *(const v16u*)&cur[(((nt_base + ni) << 5) + lane) << 4]);

#pragma unroll
        for (int mi = 0; mi < 2; ++mi)
#pragma unroll
            for (int ni = 0; ni < 4; ++ni)
                acc[mi][ni] = __builtin_amdgcn_wmma_f32_16x16x32_bf16(
                    false, afr[mi], false, bfr[ni], (short)0, acc[mi][ni],
                    false, false);

        // overlap: stage next chunk while WMMAs drain
        if (chunk < 17)
            stage_B(S, nxt, b, y, chunk + 1, tid);
    }

    __syncthreads();
    // -- spill flow tile to LDS (bf16). C/D layout: lane<16 -> M=r, else M=8+r.
    {
        int half = lane >> 4, lm = lane & 15;
#pragma unroll
        for (int mi = 0; mi < 2; ++mi) {
            int m0 = ((mt_base + mi) << 4) + (half << 3);
#pragma unroll
            for (int ni = 0; ni < 4; ++ni) {
                int n = ((nt_base + ni) << 4) + lm;
#pragma unroll
                for (int r = 0; r < 8; ++r)
                    F[(m0 + r) * 128 + n] = f2bf(acc[mi][ni][r]);
            }
        }
    }
    __syncthreads();

    // -- warp + gated blend: 64 ch x 128 px per row, gathers hit L2
    int p = tid & 127;
    float g  = gates[b * (LH_ * LW_) + y * LW_ + p];
    float bx = -1.f + 2.f * p * (1.f / 127.f);
    float by = -1.f + 2.f * y * (1.f / 127.f);
    for (int c = tid >> 7; c < C_; c += 4) {
        float fhx = bf2f(F[(2 * c)       * 128 + p]);
        float fhy = bf2f(F[(2 * c + 1)   * 128 + p]);
        float flx = bf2f(F[(128 + 2 * c) * 128 + p]);
        float fly = bf2f(F[(129 + 2 * c) * 128 + p]);
        const float* him = high + (b * C_ + c) * (HH_ * HW_);
        const float* lim = low  + (b * C_ + c) * (LH_ * LW_);
        float gxh = (bx + fhx * (1.f / 128.f) + 1.f) * 0.5f * (HW_ - 1);
        float gyh = (by + fhy * (1.f / 128.f) + 1.f) * 0.5f * (HH_ - 1);
        float gxl = (bx + flx * (1.f / 128.f) + 1.f) * 0.5f * (LW_ - 1);
        float gyl = (by + fly * (1.f / 128.f) + 1.f) * 0.5f * (LH_ - 1);
        float hv = bilin_sample(him, HH_, HW_, gxh, gyh);
        float lv = bilin_sample(lim, LH_, LW_, gxl, gyl);
        out[((b * C_ + c) * LH_ + y) * LW_ + p] = hv * g + lv * (1.f - g);
    }
}

// ---------------------------------------------------------------------------
extern "C" void kernel_launch(void* const* d_in, const int* in_sizes, int n_in,
                              void* d_out, int out_size, void* d_ws, size_t ws_size,
                              hipStream_t stream) {
    const float* low  = (const float*)d_in[0];   // (8,64,128,128)
    const float* high = (const float*)d_in[1];   // (8,64,64,64)
    const float* w_l  = (const float*)d_in[2];   // (64,1,3,3)
    const float* w_h  = (const float*)d_in[3];   // (64,1,3,3)
    const float* w_f  = (const float*)d_in[4];   // (256,64,3,3)
    const float* w_g  = (const float*)d_in[5];   // (1,4,3,3)
    float* out = (float*)d_out;

    char* ws = (char*)d_ws;
    const size_t SZ_HFEAT = (size_t)B_ * C_ * HH_ * HW_ * 4;        //  8.39 MB
    const size_t SZ_SBF   = (size_t)B_ * C_ * LH_ * LW_ * 2;        // 16.78 MB
    const size_t SZ_GIN   = (size_t)B_ * 4 * LH_ * LW_ * 4;         //  2.10 MB
    const size_t SZ_GATES = (size_t)B_ * LH_ * LW_ * 4;             //  0.52 MB
    float*          hfeat   = (float*)ws;
    unsigned short* s_bf    = (unsigned short*)(ws + SZ_HFEAT);
    float*          gate_in = (float*)(ws + SZ_HFEAT + SZ_SBF);
    float*          gatesb  = (float*)(ws + SZ_HFEAT + SZ_SBF + SZ_GIN);
    unsigned short* Abf     = (unsigned short*)(ws + SZ_HFEAT + SZ_SBF + SZ_GIN + SZ_GATES);

    k_prep_weights<<<(18 * 16 * 32 * 16) / 256, 256, 0, stream>>>(w_f, Abf);
    k_dw_high<<<(B_ * C_ * HH_ * HW_) / 256, 256, 0, stream>>>(high, w_h, hfeat);
    k_fuse_pre<<<(B_ * LH_ * LW_) / 256, 256, 0, stream>>>(low, w_l, hfeat, s_bf, gate_in);
    k_gate_conv<<<(B_ * LH_ * LW_) / 256, 256, 0, stream>>>(gate_in, w_g, gatesb);
    k_flow_wmma_warp<<<B_ * LH_, 512, 0, stream>>>(Abf, s_bf, low, high, gatesb, out);
}
